// OriginalLSTM_47442208751620
// MI455X (gfx1250) — matrix operand order repaired and checked by reading, not compiled, not run on hardware
//
#include <hip/hip_runtime.h>
#include <hip/hip_bf16.h>

// ---------------------------------------------------------------------------
// Types for CDNA5 WMMA (gfx1250, wave32)
// ---------------------------------------------------------------------------
typedef __bf16 bf16;
typedef bf16  bf16x8  __attribute__((ext_vector_type(8)));
typedef bf16  bf16x16 __attribute__((ext_vector_type(16)));
typedef float f32x8   __attribute__((ext_vector_type(8)));

// Problem constants (match reference)
#define VOCAB 32000
#define EDIM  512
#define HDIM  1024
#define GDIM  4096   // 4*H
#define BATCH 4
#define SEQ   512
#define MROWS 2048   // B*S
#define BK    32     // k-panel depth (one WMMA K per panel step)

// ---------------------------------------------------------------------------
// WMMA fragment loaders (ISA 7.12.2 layouts, wave32)
// ---------------------------------------------------------------------------

// A-matrix 16x32 bf16, row-major source with leading dim lda (elements).
// lane<16 : row=lane,    K = [0..7]  and [16..23]
// lane>=16: row=lane-16, K = [8..15] and [24..31]
__device__ __forceinline__ bf16x16 load_fragA(const bf16* base, int lda, int lane) {
    int r  = lane & 15;
    int hf = lane >> 4;
    const bf16* p = base + (size_t)r * lda + hf * 8;
    union { bf16x16 v; bf16x8 h[2]; } u;
    u.h[0] = *(const bf16x8*)(p);        // K = hf*8 .. +7
    u.h[1] = *(const bf16x8*)(p + 16);   // K = 16+hf*8 .. +7
    return u.v;
}

// B-matrix 32x16 bf16 where B[k][n] = W[n][k]; W row-major, leading dim ldb.
// lane<16 : n=lane,    K = [0..7] and [16..23] ; lane>=16: n=lane-16, K = [8..15] and [24..31]
__device__ __forceinline__ bf16x16 load_fragB(const bf16* Wrow0, int ldb, int lane) {
    int n  = lane & 15;
    int kh = lane >> 4;
    const bf16* p = Wrow0 + (size_t)n * ldb + kh * 8;
    union { bf16x16 v; bf16x8 h[2]; } u;
    u.h[0] = *(const bf16x8*)(p);
    u.h[1] = *(const bf16x8*)(p + 16);
    return u.v;
}

// ---------------------------------------------------------------------------
// CDNA5 async copy: global -> LDS, 16 bytes per lane, tracked by ASYNCcnt.
// VDST carries the per-lane LDS byte address (low 32 bits of generic ptr).
// ---------------------------------------------------------------------------
__device__ __forceinline__ void async_copy_b128(unsigned int lds_byte_addr,
                                                const void* gptr) {
    asm volatile("global_load_async_to_lds_b128 %0, %1, off"
                 :: "v"(lds_byte_addr), "v"((unsigned long long)(uintptr_t)gptr)
                 : "memory");
}
__device__ __forceinline__ void wait_asynccnt0() {
    asm volatile("s_wait_asynccnt 0x0" ::: "memory");
}

// ---------------------------------------------------------------------------
// Elementwise helpers / prep kernels
// ---------------------------------------------------------------------------
__global__ void cvt_f32_bf16_kernel(const float* __restrict__ in,
                                    bf16* __restrict__ out, size_t n) {
    size_t i = (size_t)blockIdx.x * blockDim.x + threadIdx.x;
    size_t stride = (size_t)gridDim.x * blockDim.x;
    for (; i < n; i += stride) out[i] = (bf16)in[i];
}

__global__ void bias_combine_kernel(const float* __restrict__ bih,
                                    const float* __restrict__ bhh,
                                    float* __restrict__ bc, int n) {
    int i = blockIdx.x * blockDim.x + threadIdx.x;
    if (i < n) bc[i] = bih[i] + bhh[i];
}

// Embedding gather -> bf16 code[M=2048][E=512]
__global__ void embed_kernel(const int* __restrict__ x,
                             const float* __restrict__ emb,
                             bf16* __restrict__ code) {
    int i = blockIdx.x * blockDim.x + threadIdx.x;   // over 2048*512
    if (i < MROWS * EDIM) {
        int m = i >> 9;        // /512
        int e = i & 511;
        code[i] = (bf16)emb[(size_t)x[m] * EDIM + e];
    }
}

// ---------------------------------------------------------------------------
// GEMM: C[M][N] = A[M][K] * W[N][K]^T + bias[N]    (bf16 in, f32 out)
// Block = 256 threads (8 waves). Block tile 128(M) x 128(N).
// Waves: 4(M) x 2(N); wave tile 32x64 = 8 WMMA accumulators.
// A/B k-panels (128x32 bf16 each) double-buffered in LDS, filled with
// global_load_async_to_lds_b128 (ASYNCcnt) and consumed via ds_load_b128.
// ---------------------------------------------------------------------------
__global__ __launch_bounds__(256)
void wmma_gemm_lds_kernel(const bf16* __restrict__ A,
                          const bf16* __restrict__ W,
                          const float* __restrict__ bias,
                          float* __restrict__ C,
                          int M, int N, int K) {
    __shared__ bf16 sA[2][128][BK];   // 16 KB
    __shared__ bf16 sB[2][128][BK];   // 16 KB

    const int tid  = threadIdx.x;
    const int lane = tid & 31;
    const int wave = tid >> 5;       // 0..7
    const int wy   = wave >> 1;      // 0..3 : M
    const int wx   = wave & 1;       // 0..1 : N
    const int m0g  = blockIdx.y * 128;
    const int n0g  = blockIdx.x * 128;

    // Copy assignment: 256 threads x (2 chunks A + 2 chunks B) of 16B.
    // thread covers 32 contiguous bytes of one panel row.
    const int crow = tid >> 1;              // 0..127 (panel row)
    const int cel  = (tid & 1) * 16;        // element offset 0 or 16

    const unsigned int ldsA0 = (unsigned int)(uintptr_t)(&sA[0][0][0]);
    const unsigned int ldsB0 = (unsigned int)(uintptr_t)(&sB[0][0][0]);
    const unsigned int panelBytes = 128 * BK * 2;   // 8 KB

    auto issue_panel = [&](int k0, int buf) {
        const bf16* ga = A + (size_t)(m0g + crow) * K + k0 + cel;
        const bf16* gb = W + (size_t)(n0g + crow) * K + k0 + cel;
        unsigned int la = ldsA0 + (unsigned)buf * panelBytes + crow * (BK * 2) + cel * 2;
        unsigned int lb = ldsB0 + (unsigned)buf * panelBytes + crow * (BK * 2) + cel * 2;
        async_copy_b128(la,      ga);
        async_copy_b128(la + 16, ga + 8);
        async_copy_b128(lb,      gb);
        async_copy_b128(lb + 16, gb + 8);
    };

    f32x8 acc[2][4];
#pragma unroll
    for (int i = 0; i < 2; ++i)
#pragma unroll
        for (int j = 0; j < 4; ++j)
            acc[i][j] = (f32x8){0.f,0.f,0.f,0.f,0.f,0.f,0.f,0.f};

    issue_panel(0, 0);
    wait_asynccnt0();
    __syncthreads();

    const int nk = K / BK;
    for (int kk = 0; kk < nk; ++kk) {
        const int buf = kk & 1;
        if (kk + 1 < nk) issue_panel((kk + 1) * BK, buf ^ 1);

        // Batched LDS fragment loads, then 8 WMMAs.
        bf16x16 bfrag[4];
#pragma unroll
        for (int j = 0; j < 4; ++j)
            bfrag[j] = load_fragB(&sB[buf][wx * 64 + j * 16][0], BK, lane);
#pragma unroll
        for (int i = 0; i < 2; ++i) {
            bf16x16 afrag = load_fragA(&sA[buf][wy * 32 + i * 16][0], BK, lane);
#pragma unroll
            for (int j = 0; j < 4; ++j)
                acc[i][j] = __builtin_amdgcn_wmma_f32_16x16x32_bf16(
                    false, afrag, false, bfrag[j], (short)0, acc[i][j], false, false);
        }

        wait_asynccnt0();     // next panel has landed in LDS
        __syncthreads();      // ... and is visible to all waves
    }

    // Epilogue: C/D layout row = vgpr + 8*(lane>=16), col = lane&15.
    const int rbase = (lane >> 4) * 8;
    const int col_l = lane & 15;
#pragma unroll
    for (int i = 0; i < 2; ++i) {
#pragma unroll
        for (int j = 0; j < 4; ++j) {
            int n = n0g + wx * 64 + j * 16 + col_l;
            float bv = bias[n];
#pragma unroll
            for (int r = 0; r < 8; ++r) {
                C[(size_t)(m0g + wy * 32 + i * 16 + rbase + r) * N + n] = acc[i][j][r] + bv;
            }
        }
    }
}

// ---------------------------------------------------------------------------
// LSTM recurrence: single persistent workgroup (1024 threads = 32 waves).
// h (bf16, padded to 16 rows) and c (f32) live in LDS; per step each wave
// computes gates for 32 H-columns via WMMA against L2-resident W_hh (bf16).
// xg already contains input projection + combined bias.
// ---------------------------------------------------------------------------
__global__ __launch_bounds__(1024)
void lstm_scan_kernel(const float* __restrict__ xg,     // [B][S][4H]
                      const bf16* __restrict__ Whh,     // [4H][H]
                      bf16* __restrict__ hs) {          // [B][S][H]
    __shared__ bf16  h_lds[16][HDIM];   // rows 4..15 stay zero (WMMA M padding)
    __shared__ float c_lds[BATCH][HDIM];

    const int tid  = threadIdx.x;
    const int lane = tid & 31;
    const int wave = tid >> 5;              // 0..31, owns H cols [wave*32, wave*32+32)

    for (int i = tid; i < 16 * HDIM; i += 1024) (&h_lds[0][0])[i] = (bf16)0.f;
    for (int i = tid; i < BATCH * HDIM; i += 1024) (&c_lds[0][0])[i] = 0.f;
    __syncthreads();

    const int hf    = lane >> 4;
    const int col_l = lane & 15;

    for (int t = 0; t < SEQ; ++t) {
#pragma unroll 2
        for (int tt = 0; tt < 2; ++tt) {
            const int c0 = wave * 32 + tt * 16;
            f32x8 acc[4];
#pragma unroll
            for (int g = 0; g < 4; ++g) acc[g] = (f32x8){0.f,0.f,0.f,0.f,0.f,0.f,0.f,0.f};

            for (int k0 = 0; k0 < HDIM; k0 += 32) {
                bf16x16 a = load_fragA(&h_lds[0][0] + k0, HDIM, lane);
                bf16x16 bfrag[4];
#pragma unroll
                for (int g = 0; g < 4; ++g)
                    bfrag[g] = load_fragB(Whh + (size_t)(g * HDIM + c0) * HDIM + k0,
                                          HDIM, lane);
#pragma unroll
                for (int g = 0; g < 4; ++g)
                    acc[g] = __builtin_amdgcn_wmma_f32_16x16x32_bf16(
                        false, a, false, bfrag[g], (short)0, acc[g], false, false);
            }

            if (tt == 1) __syncthreads();   // all waves done reading h this step

            const int col = c0 + col_l;
#pragma unroll
            for (int r = 0; r < 8; ++r) {
                const int row = r + 8 * hf;      // batch index (valid < 4)
                if (row < BATCH) {
                    const float* xgp = xg + ((size_t)row * SEQ + t) * GDIM;
                    float vi = acc[0][r] + xgp[0 * HDIM + col];
                    float vf = acc[1][r] + xgp[1 * HDIM + col];
                    float vg = acc[2][r] + xgp[2 * HDIM + col];
                    float vo = acc[3][r] + xgp[3 * HDIM + col];
                    float i_ = 1.f / (1.f + __expf(-vi));
                    float f_ = 1.f / (1.f + __expf(-vf));
                    float g_ = tanhf(vg);
                    float o_ = 1.f / (1.f + __expf(-vo));
                    float c_ = f_ * c_lds[row][col] + i_ * g_;
                    float h_ = o_ * tanhf(c_);
                    c_lds[row][col] = c_;
                    h_lds[row][col] = (bf16)h_;
                    hs[((size_t)row * SEQ + t) * HDIM + col] = (bf16)h_;
                }
            }
        }
        __syncthreads();   // new h visible before next step's A-fragment reads
    }
}

// ---------------------------------------------------------------------------
// Host-side orchestration
// ---------------------------------------------------------------------------
extern "C" void kernel_launch(void* const* d_in, const int* in_sizes, int n_in,
                              void* d_out, int out_size, void* d_ws, size_t ws_size,
                              hipStream_t stream) {
    const int*   x    = (const int*)  d_in[0];   // [B,S]
    const float* emb  = (const float*)d_in[1];   // [V,E]
    const float* Wih  = (const float*)d_in[2];   // [4H,E]
    const float* Whh  = (const float*)d_in[3];   // [4H,H]
    const float* bih  = (const float*)d_in[4];   // [4H]
    const float* bhh  = (const float*)d_in[5];   // [4H]
    const float* Wfc  = (const float*)d_in[6];   // [V,H]
    const float* bfc  = (const float*)d_in[7];   // [V]
    float* out = (float*)d_out;                  // [B,S,V]

    // Workspace carve (all offsets 256B aligned)
    char* ws = (char*)d_ws;
    size_t off = 0;
    bf16* Wih_bf  = (bf16*)(ws + off); off += (size_t)GDIM * EDIM * 2;        //  4 MB
    bf16* Whh_bf  = (bf16*)(ws + off); off += (size_t)GDIM * HDIM * 2;        //  8 MB
    bf16* Wfc_bf  = (bf16*)(ws + off); off += (size_t)VOCAB * HDIM * 2;       // 64 MB
    bf16* code_bf = (bf16*)(ws + off); off += (size_t)MROWS * EDIM * 2;       //  2 MB
    bf16* hs_bf   = (bf16*)(ws + off); off += (size_t)MROWS * HDIM * 2;       //  4 MB
    float* bias_c = (float*)(ws + off); off += (size_t)GDIM * 4;              // 16 KB
    float* xg     = (float*)(ws + off); off += (size_t)MROWS * GDIM * 4;      // 32 MB

    // 1) weight precision conversion (fp32 -> bf16)
    cvt_f32_bf16_kernel<<<2048, 256, 0, stream>>>(Wih, Wih_bf, (size_t)GDIM * EDIM);
    cvt_f32_bf16_kernel<<<2048, 256, 0, stream>>>(Whh, Whh_bf, (size_t)GDIM * HDIM);
    cvt_f32_bf16_kernel<<<4096, 256, 0, stream>>>(Wfc, Wfc_bf, (size_t)VOCAB * HDIM);
    bias_combine_kernel<<<(GDIM + 255) / 256, 256, 0, stream>>>(bih, bhh, bias_c, GDIM);

    // 2) embedding gather -> bf16
    embed_kernel<<<(MROWS * EDIM + 255) / 256, 256, 0, stream>>>(x, emb, code_bf);

    // 3) input projection: xg[2048][4096] = code * Wih^T + (b_ih+b_hh)
    {
        dim3 grid(GDIM / 128, MROWS / 128);   // 32 x 16
        wmma_gemm_lds_kernel<<<grid, 256, 0, stream>>>(
            code_bf, Wih_bf, bias_c, xg, MROWS, GDIM, EDIM);
    }

    // 4) sequential LSTM scan (single persistent workgroup, h/c in LDS)
    lstm_scan_kernel<<<1, 1024, 0, stream>>>(xg, Whh_bf, hs_bf);

    // 5) output projection: out[2048][32000] = hs * Wfc^T + b_fc
    {
        dim3 grid(VOCAB / 128, MROWS / 128);  // 250 x 16
        wmma_gemm_lds_kernel<<<grid, 256, 0, stream>>>(
            hs_bf, Wfc_bf, bfc, out, MROWS, VOCAB, HDIM);
    }

    (void)in_sizes; (void)n_in; (void)out_size; (void)ws_size;
}
